// DeformableTransformer_53730040873280
// MI455X (gfx1250) — matrix operand order repaired and checked by reading, not compiled, not run on hardware
//
#include <hip/hip_runtime.h>
#include <hip/hip_bf16.h>
#include <math.h>

// ---------------------------------------------------------------------------
// Problem constants (static in the reference)
// ---------------------------------------------------------------------------
#define BATCH  8
#define LQ     300
#define C_DIM  256
#define HEADS  8
#define DH_DIM 32
#define LEVELS 4
#define POINTS 4
#define DFFN   1024
#define LIN_T  19947   // sum of level H*W

typedef __attribute__((ext_vector_type(16))) __bf16       v16bf;
typedef __attribute__((ext_vector_type(8)))  float        v8f;
typedef __attribute__((ext_vector_type(4)))  float        fv4;
typedef __attribute__((ext_vector_type(4)))  unsigned int uv4;

union Frag16 {                 // 16 bf16 = 8 VGPRs = 2x 16B
    v16bf          v;
    unsigned short s[16];
    uv4            q[2];
};
union CFrag { v8f v; float e[8]; };

__device__ __forceinline__ unsigned short bf16_bits(float f)
{
    __bf16 h = (__bf16)f;
    unsigned short u;
    __builtin_memcpy(&u, &h, 2);
    return u;
}
__device__ __forceinline__ unsigned int pack2(float lo, float hi)
{
    return (unsigned int)bf16_bits(lo) | ((unsigned int)bf16_bits(hi) << 16);
}

// LDS row stride: 40 ushorts = 80B (16B aligned; row*20 dwords mod 64 banks ->
// 16 distinct bank groups across the 16 lanes of a fragment read)
#define LDSW 40

// ---------------------------------------------------------------------------
// Generic batched GEMM:  Cz = alpha * (Az @ Bz) + bias, optional relu/rowmask
//   A: [M,K] f32 row-major (lda);  B: [K,N] (ldb) or, if transB, [N,K] (ldb)
//   batch offsets: base + (z/div)*stride_outer + (z%div)*stride_inner
//   f32 -> bf16 LDS tiles -> v_wmma_f32_16x16x32_bf16 (f32 accumulate)
//   Block: 256 threads = 8 waves; tile 128(M) x 64(N); K step 32.
//   Wave = 32x32 output: 4 accumulators, 2 A-frags, 2 B-frags, 4 WMMAs/step.
//   Main K loop is branch-free for A (row clamp) and near-branch-free for B;
//   the K tail (K % 32, only the att@V GEMM) runs one fully guarded step.
// ---------------------------------------------------------------------------
__global__ __launch_bounds__(256) void gemm_bf16_kernel(
    const float* __restrict__ A, int lda, long sAo, long sAi, int dA,
    const float* __restrict__ Bm, int ldb, long sBo, long sBi, int dB, int transB,
    const float* __restrict__ bias,
    const unsigned char* __restrict__ rowmask,
    float* __restrict__ Cm, int ldc, long sCo, long sCi, int dC,
    int M, int N, int K, float alpha, int relu)
{
    __shared__ __align__(16) unsigned short sA [128][LDSW];  // [m][k] bf16
    __shared__ __align__(16) unsigned short sBT[64][LDSW];   // [n][k] bf16

    const int z = blockIdx.z;
    const float* Ab = A  + (long)(z / dA) * sAo + (long)(z % dA) * sAi;
    const float* Bb = Bm + (long)(z / dB) * sBo + (long)(z % dB) * sBi;
    float*       Cb = Cm + (long)(z / dC) * sCo + (long)(z % dC) * sCi;

    const int m0   = blockIdx.y * 128;
    const int n0   = blockIdx.x * 64;
    const int tid  = threadIdx.x;
    const int lane = tid & 31;
    const int wave = tid >> 5;
    const int hf   = (lane >> 4) & 1;   // lane half
    const int l16  = lane & 15;
    const int wm   = wave & 3;          // 32-row band 0..3
    const int wn   = wave >> 2;         // 32-col band 0..1

    // staging coordinates (row-of-8-k pattern, for A and transB-B)
    const int rowA = tid >> 2;          // 0..63
    const int kA   = (tid & 3) * 8;     // 0,8,16,24
    // staging coordinates for non-trans B: 2(k) x 4(n) micro-tile
    const int n4   = (tid & 15) * 4;    // 0..60
    const int k2   = (tid >> 4) * 2;    // 0..30

    CFrag c00, c01, c10, c11;
#pragma unroll
    for (int i = 0; i < 8; ++i) {
        c00.e[i] = 0.f; c01.e[i] = 0.f; c10.e[i] = 0.f; c11.e[i] = 0.f;
    }

    // fragment load + 4 WMMAs for one staged 128x32 * 32x64 step
    auto mma_step = [&]() {
        Frag16 a0, a1, b0, b1;
        const unsigned short* ra0 = &sA[wm * 32 + l16][0];
        const unsigned short* ra1 = &sA[wm * 32 + 16 + l16][0];
        a0.q[0] = *(const uv4*)(ra0 + 8 * hf);        // k = 8h..8h+7
        a0.q[1] = *(const uv4*)(ra0 + 16 + 8 * hf);   // k = 16+8h..
        a1.q[0] = *(const uv4*)(ra1 + 8 * hf);
        a1.q[1] = *(const uv4*)(ra1 + 16 + 8 * hf);
        const unsigned short* rb0 = &sBT[wn * 32 + l16][0];
        const unsigned short* rb1 = &sBT[wn * 32 + 16 + l16][0];
        b0.q[0] = *(const uv4*)(rb0 + 8 * hf);        // k = 8h..8h+7
        b0.q[1] = *(const uv4*)(rb0 + 16 + 8 * hf);   // k = 16+8h..
        b1.q[0] = *(const uv4*)(rb1 + 8 * hf);
        b1.q[1] = *(const uv4*)(rb1 + 16 + 8 * hf);
        c00.v = __builtin_amdgcn_wmma_f32_16x16x32_bf16(
                    false, a0.v, false, b0.v, (short)0, c00.v, false, false);
        c01.v = __builtin_amdgcn_wmma_f32_16x16x32_bf16(
                    false, a0.v, false, b1.v, (short)0, c01.v, false, false);
        c10.v = __builtin_amdgcn_wmma_f32_16x16x32_bf16(
                    false, a1.v, false, b0.v, (short)0, c10.v, false, false);
        c11.v = __builtin_amdgcn_wmma_f32_16x16x32_bf16(
                    false, a1.v, false, b1.v, (short)0, c11.v, false, false);
    };

    const int kMain = K & ~31;

    for (int k0 = 0; k0 < kMain; k0 += 32) {
        // ---- stage A 128x32: branch-free (row clamp; OOB rows never stored)
#pragma unroll
        for (int r = 0; r < 128; r += 64) {
            const int row = rowA + r;
            int gm = m0 + row;
            if (gm > M - 1) gm = M - 1;
            const float* p = Ab + (long)gm * lda + (k0 + kA);
            fv4 f0 = *(const fv4*)p;
            fv4 f1 = *(const fv4*)(p + 4);
            uv4 d;
            d.x = pack2(f0.x, f0.y); d.y = pack2(f0.z, f0.w);
            d.z = pack2(f1.x, f1.y); d.w = pack2(f1.z, f1.w);
            *(uv4*)&sA[row][kA] = d;            // ds_store_b128
        }
        // ---- stage B 32x64 into sBT[n][k]
        if (transB) {                            // B is [N,K]: rows are k-runs
            const int gn = n0 + rowA;
            uv4 d = {0u, 0u, 0u, 0u};
            if (gn < N) {
                const float* p = Bb + (long)gn * ldb + (k0 + kA);
                fv4 f0 = *(const fv4*)p;
                fv4 f1 = *(const fv4*)(p + 4);
                d.x = pack2(f0.x, f0.y); d.y = pack2(f0.z, f0.w);
                d.z = pack2(f1.x, f1.y); d.w = pack2(f1.z, f1.w);
            }
            *(uv4*)&sBT[rowA][kA] = d;
        } else {                                 // B is [K,N]
            const int gn = n0 + n4;
            if (gn + 4 <= N) {                   // 2 coalesced float4 rows
                const float* p0 = Bb + (long)(k0 + k2) * ldb + gn;
                fv4 g0 = *(const fv4*)p0;
                fv4 g1 = *(const fv4*)(p0 + ldb);
                *(unsigned int*)&sBT[n4 + 0][k2] = pack2(g0.x, g1.x);
                *(unsigned int*)&sBT[n4 + 1][k2] = pack2(g0.y, g1.y);
                *(unsigned int*)&sBT[n4 + 2][k2] = pack2(g0.z, g1.z);
                *(unsigned int*)&sBT[n4 + 3][k2] = pack2(g0.w, g1.w);
            } else {                             // N edge: guarded columns
                const float* p0 = Bb + (long)(k0 + k2) * ldb;
#pragma unroll
                for (int j = 0; j < 4; ++j) {
                    const bool ok = (gn + j) < N;
                    float a0 = ok ? p0[gn + j] : 0.f;
                    float a1 = ok ? p0[ldb + gn + j] : 0.f;
                    *(unsigned int*)&sBT[n4 + j][k2] = pack2(a0, a1);
                }
            }
        }
        __syncthreads();
        mma_step();
        __syncthreads();
    }

    // ---- K tail (K % 32 != 0): one fully guarded, zero-filled step ----
    if (kMain < K) {
        const int k0 = kMain;
#pragma unroll
        for (int r = 0; r < 128; r += 64) {
            const int row = rowA + r;
            const int gm = m0 + row;
#pragma unroll
            for (int j = 0; j < 8; ++j) {
                const int gk = k0 + kA + j;
                float v = (gm < M && gk < K) ? Ab[(long)gm * lda + gk] : 0.f;
                sA[row][kA + j] = bf16_bits(v);
            }
        }
        if (transB) {
            const int gn = n0 + rowA;
#pragma unroll
            for (int j = 0; j < 8; ++j) {
                const int gk = k0 + kA + j;
                float v = (gn < N && gk < K) ? Bb[(long)gn * ldb + gk] : 0.f;
                sBT[rowA][kA + j] = bf16_bits(v);
            }
        } else {
            const int gn = n0 + n4;
#pragma unroll
            for (int dk = 0; dk < 2; ++dk) {
#pragma unroll
                for (int j = 0; j < 4; ++j) {
                    const int gk = k0 + k2 + dk;
                    float v = (gk < K && gn + j < N)
                                  ? Bb[(long)gk * ldb + gn + j] : 0.f;
                    sBT[n4 + j][k2 + dk] = bf16_bits(v);
                }
            }
        }
        __syncthreads();
        mma_step();
        __syncthreads();
    }

    // ---- epilogue: alpha, bias, relu, rowmask, bounds-checked store ----
#pragma unroll
    for (int mt = 0; mt < 2; ++mt) {
        const CFrag& cn0 = mt ? c10 : c00;
        const CFrag& cn1 = mt ? c11 : c01;
#pragma unroll
        for (int r = 0; r < 8; ++r) {
            const int gm = m0 + wm * 32 + mt * 16 + r + 8 * hf;
            if (gm >= M) continue;
            const int gn0 = n0 + wn * 32 + l16;
            const int gn1 = gn0 + 16;
            float v0 = cn0.e[r] * alpha;
            float v1 = cn1.e[r] * alpha;
            if (bias) {
                if (gn0 < N) v0 += bias[gn0];
                if (gn1 < N) v1 += bias[gn1];
            }
            if (relu) { v0 = fmaxf(v0, 0.f); v1 = fmaxf(v1, 0.f); }
            if (rowmask && rowmask[gm]) { v0 = 0.f; v1 = 0.f; }
            if (gn0 < N) Cb[(long)gm * ldc + gn0] = v0;
            if (gn1 < N) Cb[(long)gm * ldc + gn1] = v1;
        }
    }
}

// ---------------------------------------------------------------------------
// Elementwise add
// ---------------------------------------------------------------------------
__global__ void add_kernel(const float* __restrict__ a,
                           const float* __restrict__ b,
                           float* __restrict__ o, int n)
{
    int i = blockIdx.x * blockDim.x + threadIdx.x;
    if (i < n) o[i] = a[i] + b[i];
}

// ---------------------------------------------------------------------------
// In-place row softmax, one wave32 per row of length n
// ---------------------------------------------------------------------------
__global__ __launch_bounds__(256) void softmax_rows_kernel(float* __restrict__ x,
                                                           int rows, int n)
{
    const int row  = blockIdx.x * 8 + (threadIdx.x >> 5);
    const int lane = threadIdx.x & 31;
    if (row >= rows) return;
    float* p = x + (long)row * n;
    float mx = -INFINITY;
    for (int i = lane; i < n; i += 32) mx = fmaxf(mx, p[i]);
#pragma unroll
    for (int o = 16; o; o >>= 1) mx = fmaxf(mx, __shfl_xor(mx, o, 32));
    float sum = 0.f;
    for (int i = lane; i < n; i += 32) { float e = __expf(p[i] - mx); p[i] = e; sum += e; }
#pragma unroll
    for (int o = 16; o; o >>= 1) sum += __shfl_xor(sum, o, 32);
    const float inv = 1.f / sum;
    for (int i = lane; i < n; i += 32) p[i] *= inv;
}

// In-place softmax over contiguous groups of 16 (deformable attn weights)
__global__ void softmax16_kernel(float* __restrict__ x, int groups)
{
    int g = blockIdx.x * blockDim.x + threadIdx.x;
    if (g >= groups) return;
    float* p = x + (long)g * 16;
    float mx = p[0];
#pragma unroll
    for (int i = 1; i < 16; ++i) mx = fmaxf(mx, p[i]);
    float s = 0.f;
#pragma unroll
    for (int i = 0; i < 16; ++i) { float e = __expf(p[i] - mx); p[i] = e; s += e; }
    const float inv = 1.f / s;
#pragma unroll
    for (int i = 0; i < 16; ++i) p[i] *= inv;
}

// ---------------------------------------------------------------------------
// Fused residual + LayerNorm over C=256, one 256-thread block per row
// ---------------------------------------------------------------------------
__global__ __launch_bounds__(256) void resid_ln_kernel(
    const float* __restrict__ res, const float* __restrict__ x,
    const float* __restrict__ w,   const float* __restrict__ b,
    float* __restrict__ out, int rows)
{
    __shared__ float red[8];
    const int row = blockIdx.x;
    if (row >= rows) return;
    const int c = threadIdx.x;
    const long base = (long)row * C_DIM + c;
    float v = res[base] + x[base];

    float s = v;
#pragma unroll
    for (int o = 16; o; o >>= 1) s += __shfl_xor(s, o, 32);
    if ((c & 31) == 0) red[c >> 5] = s;
    __syncthreads();
    float mean = 0.f;
#pragma unroll
    for (int i = 0; i < 8; ++i) mean += red[i];
    mean *= (1.0f / C_DIM);
    __syncthreads();

    const float d = v - mean;
    float s2 = d * d;
#pragma unroll
    for (int o = 16; o; o >>= 1) s2 += __shfl_xor(s2, o, 32);
    if ((c & 31) == 0) red[c >> 5] = s2;
    __syncthreads();
    float var = 0.f;
#pragma unroll
    for (int i = 0; i < 8; ++i) var += red[i];
    var *= (1.0f / C_DIM);

    out[base] = d * rsqrtf(var + 1e-5f) * w[c] + b[c];
}

// ---------------------------------------------------------------------------
// Multi-scale deformable sampling. One wave32 per (b,q,h); lane = channel d.
// Bilinear, zero padding, align_corners=False (grid_sample-equivalent).
// ---------------------------------------------------------------------------
__global__ __launch_bounds__(256) void msdeform_kernel(
    const float* __restrict__ value, const float* __restrict__ ref,
    const float* __restrict__ off,   const float* __restrict__ aw,
    float* __restrict__ out)
{
    const int Hs[LEVELS] = {100, 50, 25, 13};
    const int Ws[LEVELS] = {150, 75, 38, 19};
    const int Ls[LEVELS] = {0, 15000, 18750, 19700};

    const int gw   = blockIdx.x * 8 + (threadIdx.x >> 5);   // (b*Lq+q)*8 + h
    const int lane = threadIdx.x & 31;
    const int h    = gw & 7;
    const int bq   = gw >> 3;
    if (bq >= BATCH * LQ) return;
    const int b = bq / LQ;

    const float* offp = off + (long)bq * 256 + h * 32;   // + l*8 + p*2 + {x,y}
    const float* awp  = aw  + (long)bq * 128 + h * 16;   // + l*4 + p
    const float* refp = ref + (long)bq * (LEVELS * 2);
    const float* vb   = value + (long)b * LIN_T * C_DIM + h * DH_DIM + lane;

    float acc = 0.f;
#pragma unroll
    for (int l = 0; l < LEVELS; ++l) {
        const int   H = Hs[l], W = Ws[l];
        const long  sbase = (long)Ls[l] * C_DIM;
        const float rx = refp[l * 2 + 0];
        const float ry = refp[l * 2 + 1];
#pragma unroll
        for (int p = 0; p < POINTS; ++p) {
            const float ox = offp[l * 8 + p * 2 + 0];
            const float oy = offp[l * 8 + p * 2 + 1];
            const float xx = (rx + ox / (float)W) * (float)W - 0.5f;
            const float yy = (ry + oy / (float)H) * (float)H - 0.5f;
            const float x0 = floorf(xx), y0 = floorf(yy);
            const float fx = xx - x0,    fy = yy - y0;
            const int   xi = (int)x0,    yi = (int)y0;
            const float wgt = awp[l * 4 + p];
            float s = 0.f;
#pragma unroll
            for (int dy = 0; dy < 2; ++dy) {
#pragma unroll
                for (int dx = 0; dx < 2; ++dx) {
                    const int cx = xi + dx, cy = yi + dy;
                    const bool valid = (cx >= 0) && (cx < W) && (cy >= 0) && (cy < H);
                    const float cw = (dx ? fx : 1.f - fx) * (dy ? fy : 1.f - fy);
                    const int xc = cx < 0 ? 0 : (cx > W - 1 ? W - 1 : cx);
                    const int yc = cy < 0 ? 0 : (cy > H - 1 ? H - 1 : cy);
                    const float v = vb[sbase + (long)(yc * W + xc) * C_DIM];
                    s += valid ? cw * v : 0.f;
                }
            }
            acc += wgt * s;
        }
    }
    out[(long)bq * C_DIM + h * DH_DIM + lane] = acc;
}

// ---------------------------------------------------------------------------
// Host-side helper
// ---------------------------------------------------------------------------
static void launch_gemm(hipStream_t s,
                        const float* A, int lda, long sAo, long sAi, int dA,
                        const float* B, int ldb, long sBo, long sBi, int dB, int transB,
                        const float* bias, const unsigned char* rowmask,
                        float* C, int ldc, long sCo, long sCi, int dC,
                        int M, int N, int K, int batch, float alpha, int relu)
{
    dim3 grid((unsigned)((N + 63) / 64), (unsigned)((M + 127) / 128), (unsigned)batch);
    gemm_bf16_kernel<<<grid, 256, 0, s>>>(A, lda, sAo, sAi, dA,
                                          B, ldb, sBo, sBi, dB, transB,
                                          bias, rowmask,
                                          C, ldc, sCo, sCi, dC,
                                          M, N, K, alpha, relu);
}

extern "C" void kernel_launch(void* const* d_in, const int* in_sizes, int n_in,
                              void* d_out, int out_size, void* d_ws, size_t ws_size,
                              hipStream_t stream)
{
    (void)in_sizes; (void)n_in; (void)out_size; (void)ws_size;

    // ---- inputs (setup_inputs dict order; params flattened depth-first) ----
    const float* tgt   = (const float*)d_in[0];
    const float* qpos  = (const float*)d_in[1];
    const float* refpt = (const float*)d_in[2];
    const float* src   = (const float*)d_in[3];
    // d_in[4] spatial shapes, d_in[5] level starts: static constants, unused
    const unsigned char* pmask = (const unsigned char*)d_in[6];
    const float* Wq   = (const float*)d_in[7];
    const float* bq   = (const float*)d_in[8];
    const float* Wk   = (const float*)d_in[9];
    const float* bk   = (const float*)d_in[10];
    const float* Wv   = (const float*)d_in[11];
    const float* bv   = (const float*)d_in[12];
    const float* Wo   = (const float*)d_in[13];
    const float* bo   = (const float*)d_in[14];
    const float* Wval = (const float*)d_in[15];
    const float* bval = (const float*)d_in[16];
    const float* Woff = (const float*)d_in[17];
    const float* boff = (const float*)d_in[18];
    const float* Watt = (const float*)d_in[19];
    const float* batt = (const float*)d_in[20];
    const float* Wout = (const float*)d_in[21];
    const float* bout = (const float*)d_in[22];
    const float* ln1w = (const float*)d_in[23];
    const float* ln1b = (const float*)d_in[24];
    const float* ln2w = (const float*)d_in[25];
    const float* ln2b = (const float*)d_in[26];
    const float* ln3w = (const float*)d_in[27];
    const float* ln3b = (const float*)d_in[28];
    const float* W1   = (const float*)d_in[29];
    const float* b1   = (const float*)d_in[30];
    const float* W2   = (const float*)d_in[31];
    const float* b2   = (const float*)d_in[32];

    float* out = (float*)d_out;

    // ---- workspace layout (floats) ----
    const long BLqC = (long)BATCH * LQ * C_DIM;        // 614400
    float* ws    = (float*)d_ws;
    float* value = ws;                                  long o = (long)BATCH * LIN_T * C_DIM;
    float* qbuf  = ws + o;  o += BLqC;
    float* Qb    = ws + o;  o += BLqC;
    float* Kb    = ws + o;  o += BLqC;
    float* Vb    = ws + o;  o += BLqC;
    float* atto  = ws + o;  o += BLqC;
    float* proj  = ws + o;  o += BLqC;
    float* tgtA  = ws + o;  o += BLqC;
    float* tgtB  = ws + o;  o += BLqC;
    float* samp  = ws + o;  o += BLqC;
    float* offb  = ws + o;  o += BLqC;
    float* attw  = ws + o;  o += (long)BATCH * LQ * (HEADS * LEVELS * POINTS);
    float* ffn1  = ws + o;  o += (long)BATCH * LQ * DFFN;
    float* score = ws + o;  // B*H*Lq*Lq = 5,760,000

    const int M2400 = BATCH * LQ;
    const long sBH  = (long)LQ * C_DIM;       // per-batch stride of [B,Lq,C]
    const long sSc  = (long)LQ * LQ;          // per-(b,h) stride of scores
    const float invsq = 0.1767766952966369f;  // 1/sqrt(32)

    // ===== self-attention =====
    add_kernel<<<(BLqC + 255) / 256, 256, 0, stream>>>(tgt, qpos, qbuf, (int)BLqC);

    launch_gemm(stream, qbuf, C_DIM, 0, 0, 1, Wq, C_DIM, 0, 0, 1, 0, bq, nullptr,
                Qb, C_DIM, 0, 0, 1, M2400, C_DIM, C_DIM, 1, 1.f, 0);
    launch_gemm(stream, qbuf, C_DIM, 0, 0, 1, Wk, C_DIM, 0, 0, 1, 0, bk, nullptr,
                Kb, C_DIM, 0, 0, 1, M2400, C_DIM, C_DIM, 1, 1.f, 0);
    launch_gemm(stream, tgt,  C_DIM, 0, 0, 1, Wv, C_DIM, 0, 0, 1, 0, bv, nullptr,
                Vb, C_DIM, 0, 0, 1, M2400, C_DIM, C_DIM, 1, 1.f, 0);

    // scores[b,h] = (Qh @ Kh^T) / sqrt(dh)   (batch z = b*8+h)
    launch_gemm(stream, Qb, C_DIM, sBH, DH_DIM, HEADS,
                Kb, C_DIM, sBH, DH_DIM, HEADS, /*transB=*/1,
                nullptr, nullptr,
                score, LQ, sSc, 0, 1,
                LQ, LQ, DH_DIM, BATCH * HEADS, invsq, 0);

    softmax_rows_kernel<<<(BATCH * HEADS * LQ) / 8, 256, 0, stream>>>(
        score, BATCH * HEADS * LQ, LQ);

    // attout[b,:,h*32:] = att @ Vh
    launch_gemm(stream, score, LQ, sSc, 0, 1,
                Vb, C_DIM, sBH, DH_DIM, HEADS, 0,
                nullptr, nullptr,
                atto, C_DIM, sBH, DH_DIM, HEADS,
                LQ, DH_DIM, LQ, BATCH * HEADS, 1.f, 0);

    launch_gemm(stream, atto, C_DIM, 0, 0, 1, Wo, C_DIM, 0, 0, 1, 0, bo, nullptr,
                proj, C_DIM, 0, 0, 1, M2400, C_DIM, C_DIM, 1, 1.f, 0);

    resid_ln_kernel<<<M2400, 256, 0, stream>>>(tgt, proj, ln2w, ln2b, tgtA, M2400);

    // ===== deformable cross-attention =====
    add_kernel<<<(BLqC + 255) / 256, 256, 0, stream>>>(tgtA, qpos, qbuf, (int)BLqC);

    // value = mask(src @ Wval + bval)   -- the big memory-bound GEMM
    launch_gemm(stream, src, C_DIM, 0, 0, 1, Wval, C_DIM, 0, 0, 1, 0, bval, pmask,
                value, C_DIM, 0, 0, 1, BATCH * LIN_T, C_DIM, C_DIM, 1, 1.f, 0);

    launch_gemm(stream, qbuf, C_DIM, 0, 0, 1, Woff, 256, 0, 0, 1, 0, boff, nullptr,
                offb, 256, 0, 0, 1, M2400, 256, C_DIM, 1, 1.f, 0);
    launch_gemm(stream, qbuf, C_DIM, 0, 0, 1, Watt, 128, 0, 0, 1, 0, batt, nullptr,
                attw, 128, 0, 0, 1, M2400, 128, C_DIM, 1, 1.f, 0);

    softmax16_kernel<<<(BATCH * LQ * HEADS + 255) / 256, 256, 0, stream>>>(
        attw, BATCH * LQ * HEADS);

    msdeform_kernel<<<(BATCH * LQ * HEADS) / 8, 256, 0, stream>>>(
        value, refpt, offb, attw, samp);

    launch_gemm(stream, samp, C_DIM, 0, 0, 1, Wout, C_DIM, 0, 0, 1, 0, bout, nullptr,
                proj, C_DIM, 0, 0, 1, M2400, C_DIM, C_DIM, 1, 1.f, 0);

    resid_ln_kernel<<<M2400, 256, 0, stream>>>(tgtA, proj, ln1w, ln1b, tgtB, M2400);

    // ===== FFN =====
    launch_gemm(stream, tgtB, C_DIM, 0, 0, 1, W1, DFFN, 0, 0, 1, 0, b1, nullptr,
                ffn1, DFFN, 0, 0, 1, M2400, DFFN, C_DIM, 1, 1.f, /*relu=*/1);
    launch_gemm(stream, ffn1, DFFN, 0, 0, 1, W2, C_DIM, 0, 0, 1, 0, b2, nullptr,
                proj, C_DIM, 0, 0, 1, M2400, C_DIM, DFFN, 1, 1.f, 0);

    resid_ln_kernel<<<M2400, 256, 0, stream>>>(tgtB, proj, ln3w, ln3b, out, M2400);
}
